// EuclideanDistance_15685220565421
// MI455X (gfx1250) — 8-bit matrix operand order repaired and checked by reading, compiled, not run on hardware
//
#include <hip/hip_runtime.h>
#include <stdint.h>

typedef int v8i __attribute__((ext_vector_type(8)));
typedef int v4i __attribute__((ext_vector_type(4)));
typedef unsigned int v4u __attribute__((ext_vector_type(4)));

#define B_DIM 1024
#define D_DIM 512
#define N_DIM 16384
#define L_PAD 32
#define QMAX  127.0f

#define GLOBAL_AS __attribute__((address_space(1)))
#define LDS_AS    __attribute__((address_space(3)))

#if defined(__has_builtin)
#if __has_builtin(__builtin_amdgcn_global_load_async_to_lds_b128)
#define HAVE_ASYNC_LDS 1
#endif
#if __has_builtin(__builtin_amdgcn_s_wait_asynccnt)
#define HAVE_WAIT_ASYNC 1
#endif
#endif

#define WS_AMAXX 0
#define WS_AMAXW 4
#define WS_SCAL  64
#define WS_BIAS  128
#define WS_QB    (WS_BIAS + N_DIM * 4)
#define WS_XQ    (WS_QB + N_DIM * 4)
#define WS_WQT   (WS_XQ + B_DIM * D_DIM)

__global__ void k_init(unsigned* amx, unsigned* amw) {
    if (threadIdx.x == 0) { *amx = 0u; *amw = 0u; }
}

__global__ __launch_bounds__(256) void k_absmax_x(const float* __restrict__ x,
                                                  int n, unsigned* __restrict__ amx) {
    __shared__ float sm[256];
    float m = 0.0f;
    for (int i = blockIdx.x * blockDim.x + threadIdx.x; i < n; i += gridDim.x * blockDim.x)
        m = fmaxf(m, fabsf(x[i]));
    sm[threadIdx.x] = m;
    __syncthreads();
    for (int s = 128; s > 0; s >>= 1) {
        if (threadIdx.x < (unsigned)s) sm[threadIdx.x] = fmaxf(sm[threadIdx.x], sm[threadIdx.x + s]);
        __syncthreads();
    }
    if (threadIdx.x == 0) atomicMax(amx, __float_as_uint(sm[0]));
}

__global__ __launch_bounds__(256) void k_bias(const float* __restrict__ w,
                                              float* __restrict__ bias,
                                              unsigned* __restrict__ amw) {
    const int n = blockIdx.x * blockDim.x + threadIdx.x;
    float s = 0.0f, m = 0.0f;
    for (int d = 0; d < D_DIM; ++d) {
        float v = w[d * N_DIM + n];
        s += v * v;
        m = fmaxf(m, fabsf(v));
    }
    float b = s * (1.0f / (float)L_PAD);
    bias[n] = b;
    m = fmaxf(m, fabsf(b));
    __shared__ float sm[256];
    sm[threadIdx.x] = m;
    __syncthreads();
    for (int st = 128; st > 0; st >>= 1) {
        if (threadIdx.x < (unsigned)st) sm[threadIdx.x] = fmaxf(sm[threadIdx.x], sm[threadIdx.x + st]);
        __syncthreads();
    }
    if (threadIdx.x == 0) atomicMax(amw, __float_as_uint(sm[0]));
}

__global__ void k_finalize(const unsigned* amx, const unsigned* amw, float* scal) {
    if (threadIdx.x == 0) {
        float ax = fmaxf(__uint_as_float(*amx), 1.0f);
        float sx = fmaxf(ax / QMAX, 1e-12f);
        float sw = fmaxf(__uint_as_float(*amw) / QMAX, 1e-12f);
        scal[0] = sx;
        scal[1] = sw;
        scal[2] = sx * sw;
        scal[3] = rintf(1.0f / sx);
    }
}

__device__ __forceinline__ uint32_t pack4(float a, float b, float c, float d, float inv) {
    int q0 = (int)rintf(a * inv), q1 = (int)rintf(b * inv);
    int q2 = (int)rintf(c * inv), q3 = (int)rintf(d * inv);
    return (uint32_t)(q0 & 255) | ((uint32_t)(q1 & 255) << 8) |
           ((uint32_t)(q2 & 255) << 16) | ((uint32_t)(q3 & 255) << 24);
}

__global__ __launch_bounds__(256) void k_quant_x(const float* __restrict__ x,
                                                 const float* __restrict__ scal,
                                                 uint32_t* __restrict__ XqP) {
    const int i = blockIdx.x * blockDim.x + threadIdx.x;
    const float inv = 1.0f / scal[0];
    const float* p = x + i * 4;
    XqP[i] = pack4(p[0], p[1], p[2], p[3], inv);
}

__global__ __launch_bounds__(256) void k_quant_w(const float* __restrict__ w,
                                                 const float* __restrict__ scal,
                                                 uint32_t* __restrict__ WqT) {
    __shared__ uint32_t tile[32][33];
    const int g0 = blockIdx.x * 32;
    const int n0 = blockIdx.y * 32;
    const float inv = 1.0f / scal[1];
    const int t = threadIdx.x;
    {
        const int tn = t & 31;
        const int tg = t >> 5;
#pragma unroll
        for (int r = 0; r < 4; ++r) {
            const int g = tg + r * 8;
            const float* p = w + (g0 + g) * 4 * N_DIM + (n0 + tn);
            tile[g][tn] = pack4(p[0], p[N_DIM], p[2 * N_DIM], p[3 * N_DIM], inv);
        }
    }
    __syncthreads();
    {
        const int og = t & 31;
        const int on = t >> 5;
#pragma unroll
        for (int r = 0; r < 4; ++r) {
            const int n = on + r * 8;
            WqT[(n0 + n) * (D_DIM / 4) + g0 + og] = tile[og][n];
        }
    }
}

__global__ __launch_bounds__(256) void k_qb(const float* __restrict__ bias,
                                            const float* __restrict__ scal,
                                            int* __restrict__ qb) {
    const int n = blockIdx.x * blockDim.x + threadIdx.x;
    qb[n] = (int)rintf(bias[n] * (1.0f / scal[1]));
}

__device__ __forceinline__ void async_cp16(const uint8_t* g, uint8_t* l) {
#ifdef HAVE_ASYNC_LDS
    __builtin_amdgcn_global_load_async_to_lds_b128(
        (GLOBAL_AS v4i*)g, (LDS_AS v4i*)l, 0, 0);
#else
    *(v4u*)l = *(const v4u*)g;
#endif
}
__device__ __forceinline__ void async_wait0() {
#ifdef HAVE_ASYNC_LDS
#ifdef HAVE_WAIT_ASYNC
    __builtin_amdgcn_s_wait_asynccnt(0);
#else
    asm volatile("s_wait_asynccnt 0" ::: "memory");
#endif
#endif
}

__global__ __launch_bounds__(256, 2) void k_gemm(const uint8_t* __restrict__ Xq,
                                                 const uint8_t* __restrict__ Wq,
                                                 const int* __restrict__ qb,
                                                 const float* __restrict__ scal,
                                                 float* __restrict__ out) {
    __shared__ __align__(16) uint8_t ldsA[2][128 * 64];
    __shared__ __align__(16) uint8_t ldsB[2][128 * 64];

    const int tid  = threadIdx.x;
    const int lane = tid & 31;
    const int wid  = tid >> 5;
    const int wm = wid & 1;
    const int wn = wid >> 1;
    const int hi = lane >> 4;
    const int ln = lane & 15;
    const int tileM = blockIdx.x * 128;
    const int tileN = blockIdx.y * 128;

    const int srow = tid >> 2;
    const int sch  = (tid & 3) * 16;
    const uint8_t* gA = Xq + (size_t)(tileM + srow) * (D_DIM) + sch;
    const uint8_t* gB = Wq + (size_t)(tileN + srow) * (D_DIM) + sch;

    v8i acc[4][2];
#pragma unroll
    for (int mb = 0; mb < 4; ++mb)
#pragma unroll
        for (int nb = 0; nb < 2; ++nb)
#pragma unroll
            for (int r = 0; r < 8; ++r) acc[mb][nb][r] = 0;

    {
        async_cp16(gA,                 &ldsA[0][srow * 64 + sch]);
        async_cp16(gA + 64 * D_DIM,    &ldsA[0][(srow + 64) * 64 + sch]);
        async_cp16(gB,                 &ldsB[0][srow * 64 + sch]);
        async_cp16(gB + 64 * D_DIM,    &ldsB[0][(srow + 64) * 64 + sch]);
    }

#pragma unroll
    for (int ks = 0; ks < 8; ++ks) {
        const int buf = ks & 1;
        async_wait0();
        __syncthreads();

        if (ks < 7) {
            const int nb_ = buf ^ 1;
            const int ko = (ks + 1) * 64;
            async_cp16(gA + ko,              &ldsA[nb_][srow * 64 + sch]);
            async_cp16(gA + ko + 64 * D_DIM, &ldsA[nb_][(srow + 64) * 64 + sch]);
            async_cp16(gB + ko,              &ldsB[nb_][srow * 64 + sch]);
            async_cp16(gB + ko + 64 * D_DIM, &ldsB[nb_][(srow + 64) * 64 + sch]);
        }

        v8i a[4];
#pragma unroll
        for (int mb = 0; mb < 4; ++mb) {
            const uint8_t* pA = &ldsA[buf][(wm * 64 + mb * 16 + ln) * 64 + hi * 16];
            union { v8i v; uint64_t q[4]; } Af;
            Af.q[0] = *(const uint64_t*)(pA);
            Af.q[1] = *(const uint64_t*)(pA + 8);
            Af.q[2] = *(const uint64_t*)(pA + 32);
            Af.q[3] = *(const uint64_t*)(pA + 40);
            a[mb] = Af.v;
        }
        v8i b[2];
#pragma unroll
        for (int nb = 0; nb < 2; ++nb) {
            const uint8_t* pB = &ldsB[buf][(wn * 32 + nb * 16 + ln) * 64 + hi * 16];
            union { v8i v; v4u q[2]; } Bf;
            Bf.q[0] = *(const v4u*)(pB);
            Bf.q[1] = *(const v4u*)(pB + 32);
            b[nb] = Bf.v;
        }

#pragma unroll
        for (int mb = 0; mb < 4; ++mb)
#pragma unroll
            for (int nb = 0; nb < 2; ++nb)
                acc[mb][nb] = __builtin_amdgcn_wmma_i32_16x16x64_iu8(
                     true, a[mb],  true, b[nb], acc[mb][nb],
                     false,  false);

        __syncthreads();
    }

    const float scale = scal[2];
    const int q1 = (int)scal[3];
#pragma unroll
    for (int nb = 0; nb < 2; ++nb) {
        const int col = tileN + wn * 32 + nb * 16 + ln;
        const int extra = L_PAD * q1 * qb[col];
#pragma unroll
        for (int mb = 0; mb < 4; ++mb) {
            const int row0 = tileM + wm * 64 + mb * 16 + hi * 8;
#pragma unroll
            for (int r = 0; r < 8; ++r)
                out[(size_t)(row0 + r) * N_DIM + col] = scale * (float)(acc[mb][nb][r] + extra);
        }
    }
}

extern "C" void kernel_launch(void* const* d_in, const int* in_sizes, int n_in,
                              void* d_out, int out_size, void* d_ws, size_t ws_size,
                              hipStream_t stream) {
    const float* x = (const float*)d_in[0];
    const float* w = (const float*)d_in[1];
    (void)in_sizes; (void)n_in; (void)out_size; (void)ws_size;

    char* ws = (char*)d_ws;
    unsigned* amx  = (unsigned*)(ws + WS_AMAXX);
    unsigned* amw  = (unsigned*)(ws + WS_AMAXW);
    float*    scal = (float*)(ws + WS_SCAL);
    float*    bias = (float*)(ws + WS_BIAS);
    int*      qb   = (int*)(ws + WS_QB);
    uint32_t* XqP  = (uint32_t*)(ws + WS_XQ);
    uint32_t* WqT  = (uint32_t*)(ws + WS_WQT);
    float*    out  = (float*)d_out;

    k_init<<<1, 32, 0, stream>>>(amx, amw);
    k_absmax_x<<<256, 256, 0, stream>>>(x, B_DIM * D_DIM, amx);
    k_bias<<<N_DIM / 256, 256, 0, stream>>>(w, bias, amw);
    k_finalize<<<1, 32, 0, stream>>>(amx, amw, scal);
    k_quant_x<<<(B_DIM * (D_DIM / 4)) / 256, 256, 0, stream>>>(x, scal, XqP);
    {
        dim3 gq(4, 512);
        k_quant_w<<<gq, 256, 0, stream>>>(w, scal, WqT);
    }
    k_qb<<<N_DIM / 256, 256, 0, stream>>>(bias, scal, qb);

    dim3 grid(B_DIM / 128, N_DIM / 128);
    k_gemm<<<grid, 256, 0, stream>>>((const uint8_t*)XqP, (const uint8_t*)WqT, qb, scal, out);
}
